// RelationalPolicyHead_66589172957517
// MI455X (gfx1250) — compile-verified, matrix operand order checked by reading, not run on hardware
//
#include <hip/hip_runtime.h>
#include <hip/hip_bf16.h>

typedef __attribute__((ext_vector_type(16))) _Float16 v16h;
typedef __attribute__((ext_vector_type(8)))  _Float16 v8h;
typedef __attribute__((ext_vector_type(8)))  float    v8f;
typedef __attribute__((ext_vector_type(4)))  float    v4f;

#define HID   256
#define KDIM  512
#define NT    16          // 256 / 16 column tiles
#define NGRAPH 512
#define WAVES_PER_WG 8
#define MOVES_PER_WAVE 16
#define MOVES_PER_WG (WAVES_PER_WG * MOVES_PER_WAVE)   // 128
#define BCOL_STRIDE 40    // halfs per LDS column (80B) -> conflict-free ds_read_b128
#define NK32 (KDIM / 32)  // 16 K-chunks

// ---------------------------------------------------------------------------
// Pre-pass: W1 [512,256] f32 row-major  ->  W1t [256,512] f16 (col-major-k)
// ---------------------------------------------------------------------------
__global__ void w1_transpose_kernel(const float* __restrict__ W1,
                                    _Float16* __restrict__ W1t) {
    int idx = blockIdx.x * blockDim.x + threadIdx.x;
    if (idx < KDIM * HID) {
        int col = idx / KDIM;        // 0..255
        int k   = idx % KDIM;        // 0..511
        W1t[idx] = (_Float16)W1[(size_t)k * HID + col];
    }
}

// ---------------------------------------------------------------------------
// Fused gather + GEMM + ReLU + dot(W2): logits[e]
// One wave = 16 moves, full N=256 in 16 x v8f accumulators.
// W1t chunks staged into a double-buffered LDS tile with CDNA5 async-to-LDS
// copies (ASYNCcnt) overlapping the WMMA chain; B operands double-buffered
// in registers (even/odd) so each WMMA overlaps the next tile's ds_loads.
// ---------------------------------------------------------------------------
__global__ __launch_bounds__(256) void fused_logits_kernel(
    const float* __restrict__ emb,      // [N_NODES, 256]
    const int*   __restrict__ moves,    // [2, E] (src row then tgt row)
    const _Float16* __restrict__ W1t,   // [256, 512] f16
    const float* __restrict__ b1,       // [256]
    const float* __restrict__ W2,       // [256]
    const float* __restrict__ b2,       // [1]
    float* __restrict__ logits,         // [E]
    int E)
{
    __shared__ _Float16 Bs[2][HID * BCOL_STRIDE];   // 2 x 20 KB

    const int tid  = threadIdx.x;
    const int wave = tid >> 5;
    const int lane = tid & 31;
    const int lr   = lane & 15;       // row (A) / col (B) within tile
    const int hi   = lane >> 4;       // 0: K 0-7/16-23 ; 1: K 8-15/24-31
    const int koff = hi * 8;

    const int m0 = (blockIdx.x * WAVES_PER_WG + wave) * MOVES_PER_WAVE;
    const bool active = (m0 < E);

    const float* pS = emb;
    const float* pT = emb;
    if (active) {
        int m = m0 + lr;
        int s = moves[m];
        int t = moves[E + m];
        pS = emb + (size_t)s * HID;
        pT = emb + (size_t)t * HID;
    }

    v8f acc[NT];
#pragma unroll
    for (int i = 0; i < NT; ++i) acc[i] = (v8f)0.0f;

    // async stage of one 32-K W1t chunk: thread tid copies column tid (64B)
    auto stage = [&](int k32, int buf) {
        const _Float16* g = W1t + (size_t)tid * KDIM + k32 * 32;
        unsigned lds = (unsigned)(uintptr_t)(&Bs[buf][tid * BCOL_STRIDE]);
        asm volatile(
            "global_load_async_to_lds_b128 %0, %1, off\n\t"
            "global_load_async_to_lds_b128 %2, %3, off\n\t"
            "global_load_async_to_lds_b128 %4, %5, off\n\t"
            "global_load_async_to_lds_b128 %6, %7, off"
            :: "v"(lds),      "v"((unsigned long long)(g)),
               "v"(lds + 16), "v"((unsigned long long)(g + 8)),
               "v"(lds + 32), "v"((unsigned long long)(g + 16)),
               "v"(lds + 48), "v"((unsigned long long)(g + 24))
            : "memory");
    };

    auto loadB = [&](const _Float16* bbase, int nt) -> v16h {
        const _Float16* bp = bbase + nt * 16 * BCOL_STRIDE;
        v8h blo = *(const v8h*)(bp + koff);
        v8h bhiv = *(const v8h*)(bp + 16 + koff);
        v16h b;
#pragma unroll
        for (int j = 0; j < 8; ++j) { b[j] = blo[j]; b[8 + j] = bhiv[j]; }
        return b;
    };

    stage(0, 0);   // pipeline prologue

    for (int k32 = 0; k32 < NK32; ++k32) {
        // drain this wave's async stores, then publish the buffer WG-wide
        asm volatile("s_wait_asynccnt 0x0" ::: "memory");
        __syncthreads();
        const int buf = k32 & 1;

        if (k32 + 1 < NK32) stage(k32 + 1, buf ^ 1);  // overlaps compute

        if (active) {
            // ---- A operand: features [kb, kb+32) of row lr, f32 -> f16 ----
            const int kb = k32 * 32;
            const float* base = (k32 < (KDIM / 64)) ? (pS + kb) : (pT + kb - HID);
            v4f f0 = *(const v4f*)(base + koff);
            v4f f1 = *(const v4f*)(base + koff + 4);
            v4f f2 = *(const v4f*)(base + 16 + koff);
            v4f f3 = *(const v4f*)(base + 16 + koff + 4);
            v16h a;
#pragma unroll
            for (int j = 0; j < 4; ++j) {
                a[j]      = (_Float16)f0[j];
                a[4 + j]  = (_Float16)f1[j];
                a[8 + j]  = (_Float16)f2[j];
                a[12 + j] = (_Float16)f3[j];
            }

            // ---- even/odd register double-buffer for B ----
            const _Float16* bbase = &Bs[buf][lr * BCOL_STRIDE];
            v16h bA = loadB(bbase, 0);
            v16h bB = loadB(bbase, 1);
#pragma unroll
            for (int nt = 0; nt < NT; nt += 2) {
                acc[nt] = __builtin_amdgcn_wmma_f32_16x16x32_f16(
                    false, a, false, bA, (short)0, acc[nt], false, false);
                if (nt + 2 < NT) bA = loadB(bbase, nt + 2);
                acc[nt + 1] = __builtin_amdgcn_wmma_f32_16x16x32_f16(
                    false, a, false, bB, (short)0, acc[nt + 1], false, false);
                if (nt + 3 < NT) bB = loadB(bbase, nt + 3);
            }
        }
    }

    if (!active) return;

    // ---- epilogue: +b1, ReLU, *W2, reduce over the 256 columns ----
    float part[8];
#pragma unroll
    for (int v = 0; v < 8; ++v) part[v] = 0.0f;
#pragma unroll
    for (int nt = 0; nt < NT; ++nt) {
        const int col = nt * 16 + lr;
        const float bb = b1[col];
        const float ww = W2[col];
#pragma unroll
        for (int v = 0; v < 8; ++v) {
            float h = acc[nt][v] + bb;
            h = h > 0.0f ? h : 0.0f;
            part[v] += h * ww;
        }
    }
    // rows v + 8*hi live across the 16 lanes of each half: xor-reduce
#pragma unroll
    for (int v = 0; v < 8; ++v) {
        float x = part[v];
        x += __shfl_xor(x, 1);
        x += __shfl_xor(x, 2);
        x += __shfl_xor(x, 4);
        x += __shfl_xor(x, 8);
        part[v] = x;
    }
    if (lr == 0) {
        const float bias2 = b2[0];
#pragma unroll
        for (int v = 0; v < 8; ++v)
            logits[m0 + hi * 8 + v] = part[v] + bias2;
    }
}

// ---------------------------------------------------------------------------
// Segment softmax helpers (ordered-uint float max trick)
// ---------------------------------------------------------------------------
__device__ __forceinline__ unsigned f2ord(float f) {
    unsigned b = __float_as_uint(f);
    return (b & 0x80000000u) ? ~b : (b | 0x80000000u);
}
__device__ __forceinline__ float ord2f(unsigned u) {
    unsigned b = (u & 0x80000000u) ? (u & 0x7FFFFFFFu) : ~u;
    return __uint_as_float(b);
}

__global__ void seg_init_kernel(unsigned* __restrict__ segmax,
                                float* __restrict__ segsum) {
    int g = blockIdx.x * blockDim.x + threadIdx.x;
    if (g < NGRAPH) { segmax[g] = 0u; segsum[g] = 0.0f; }
}

__global__ void seg_max_kernel(const float* __restrict__ logits,
                               const int* __restrict__ moves,
                               const int* __restrict__ batch,
                               unsigned* __restrict__ segmax, int E) {
    int e = blockIdx.x * blockDim.x + threadIdx.x;
    if (e < E) {
        int g = batch[moves[e]];
        atomicMax(&segmax[g], f2ord(logits[e]));
    }
}

__global__ void seg_exp_sum_kernel(float* __restrict__ logits_ex,
                                   const int* __restrict__ moves,
                                   const int* __restrict__ batch,
                                   const unsigned* __restrict__ segmax,
                                   float* __restrict__ segsum, int E) {
    int e = blockIdx.x * blockDim.x + threadIdx.x;
    if (e < E) {
        int g = batch[moves[e]];
        float m = ord2f(segmax[g]);
        float ex = expf(logits_ex[e] - m);
        logits_ex[e] = ex;               // overwrite logits with exp values
        atomicAdd(&segsum[g], ex);
    }
}

__global__ void seg_norm_kernel(const float* __restrict__ ex,
                                const int* __restrict__ moves,
                                const int* __restrict__ batch,
                                const float* __restrict__ segsum,
                                float* __restrict__ out, int E) {
    int e = blockIdx.x * blockDim.x + threadIdx.x;
    if (e < E) {
        int g = batch[moves[e]];
        out[e] = ex[e] / (segsum[g] + 1e-16f);
    }
}

// ---------------------------------------------------------------------------
extern "C" void kernel_launch(void* const* d_in, const int* in_sizes, int n_in,
                              void* d_out, int out_size, void* d_ws, size_t ws_size,
                              hipStream_t stream) {
    const float* emb   = (const float*)d_in[0];
    const int*   moves = (const int*)d_in[1];   // [2, E]
    const int*   batch = (const int*)d_in[2];   // [N_NODES]
    const float* W1    = (const float*)d_in[3];
    const float* b1    = (const float*)d_in[4];
    const float* W2    = (const float*)d_in[5];
    const float* b2    = (const float*)d_in[6];
    float* out = (float*)d_out;
    const int E = in_sizes[1] / 2;

    char* ws = (char*)d_ws;
    size_t off = 0;
    _Float16* W1t = (_Float16*)(ws + off);
    off += (size_t)HID * KDIM * sizeof(_Float16);          // 256 KB
    float* logits = (float*)(ws + off);
    off += (size_t)E * sizeof(float);                      // 2 MB
    off = (off + 255) & ~(size_t)255;
    unsigned* segmax = (unsigned*)(ws + off);
    off += NGRAPH * sizeof(unsigned);
    float* segsum = (float*)(ws + off);

    // 1) W1 -> W1t (f16, transposed)
    w1_transpose_kernel<<<(KDIM * HID + 255) / 256, 256, 0, stream>>>(W1, W1t);
    // 2) init segment stats
    seg_init_kernel<<<(NGRAPH + 255) / 256, 256, 0, stream>>>(segmax, segsum);
    // 3) fused gather + MLP -> logits
    int nWG = (E + MOVES_PER_WG - 1) / MOVES_PER_WG;
    fused_logits_kernel<<<nWG, 256, 0, stream>>>(emb, moves, W1t, b1, W2, b2,
                                                 logits, E);
    // 4) segment softmax
    int nb = (E + 255) / 256;
    seg_max_kernel<<<nb, 256, 0, stream>>>(logits, moves, batch, segmax, E);
    seg_exp_sum_kernel<<<nb, 256, 0, stream>>>(logits, moves, batch, segmax,
                                               segsum, E);
    seg_norm_kernel<<<nb, 256, 0, stream>>>(logits, moves, batch, segsum, out, E);
}
